// HybridGATLSTM_8693013807250
// MI455X (gfx1250) — matrix operand order repaired and checked by reading, not compiled, not run on hardware
//
#include <hip/hip_runtime.h>
#include <math.h>

// Problem constants (from reference)
#define BDIM   8
#define SDIM   32
#define NNODE  2000
#define NEDGE  64000
#define ETOT   (NEDGE + NNODE)   // with self loops = 66000
#define GH     32
#define HDIM   64
#define NG     256                // 4*H gate count
#define BS     256                // B*S
#define DK     64000              // N*GH, GEMM K dim
#define NEG_SLOPE 0.2f

typedef __attribute__((ext_vector_type(16))) _Float16 v16h;
typedef __attribute__((ext_vector_type(8)))  float    v8f;

// ---------------- helpers ----------------
__device__ inline void atomicMaxF(float* addr, float val) {
  int* ai = (int*)addr;
  int old = __float_as_int(*addr);
  while (__int_as_float(old) < val) {
    int assumed = old;
    old = atomicCAS(ai, assumed, __float_as_int(val));
    if (old == assumed) break;
  }
}

__device__ inline float sigf(float x) { return 1.0f / (1.0f + expf(-x)); }

// ---------------- stage 0: scalar attention coefficients ----------------
__global__ void k_scalars(const float* __restrict__ wl, const float* __restrict__ asrc,
                          const float* __restrict__ adst, float* __restrict__ csd) {
  if (threadIdx.x == 0) {
    float cs = 0.f, cd = 0.f;
    for (int g = 0; g < GH; ++g) { cs += wl[g] * asrc[g]; cd += wl[g] * adst[g]; }
    csd[0] = cs; csd[1] = cd;
  }
}

// ---------------- stage 0b: init workspace ----------------
__global__ void k_init(float* __restrict__ segmax, float* __restrict__ denom,
                       float* __restrict__ numer, float* __restrict__ gatesx,
                       const float* __restrict__ bih, const float* __restrict__ bhh) {
  int i = blockIdx.x * 256 + threadIdx.x;
  if (i < NNODE * BS) { segmax[i] = -3.0e38f; denom[i] = 0.f; numer[i] = 0.f; }
  if (i < BS * NG)    { gatesx[i] = bih[i & 255] + bhh[i & 255]; }
}

// ---------------- stage 1: edge softmax, pass 1 (segment max) ----------------
__global__ __launch_bounds__(256)
void k_edge_max(const float* __restrict__ x, const int* __restrict__ ei,
                const float* __restrict__ csd, float* __restrict__ segmax) {
  int et = blockIdx.x;
  int srcn, dstn;
  if (et < NEDGE) { srcn = ei[et]; dstn = ei[NEDGE + et]; }
  else            { srcn = dstn = et - NEDGE; }
  int bs = threadIdx.x;
  float t = csd[0] * x[bs * NNODE + srcn] + csd[1] * x[bs * NNODE + dstn];
  float e = (t > 0.f) ? t : NEG_SLOPE * t;
  atomicMaxF(&segmax[dstn * BS + bs], e);
}

// ---------------- stage 1: pass 2 (exp, denom & numer accumulate) ----------------
__global__ __launch_bounds__(256)
void k_edge_acc(const float* __restrict__ x, const int* __restrict__ ei,
                const float* __restrict__ csd, const float* __restrict__ segmax,
                float* __restrict__ denom, float* __restrict__ numer) {
  int et = blockIdx.x;
  int srcn, dstn;
  if (et < NEDGE) { srcn = ei[et]; dstn = ei[NEDGE + et]; }
  else            { srcn = dstn = et - NEDGE; }
  int bs = threadIdx.x;
  float xs = x[bs * NNODE + srcn];
  float t = csd[0] * xs + csd[1] * x[bs * NNODE + dstn];
  float e = (t > 0.f) ? t : NEG_SLOPE * t;
  float ev = expf(e - segmax[dstn * BS + bs]);
  atomicAdd(&denom[dstn * BS + bs], ev);
  atomicAdd(&numer[dstn * BS + bs], ev * xs);
}

// ---------------- stage 1: pass 3 (agg = numer/denom, layout [N][BS]) ----------------
__global__ void k_agg(const float* __restrict__ numer, const float* __restrict__ denom,
                      float* __restrict__ agg) {
  int i = blockIdx.x * 256 + threadIdx.x;
  if (i < NNODE * BS) agg[i] = numer[i] / denom[i];
}

// ---------------- stage 2: WMMA GEMM gates_x = lstm_in @ W_ih^T ----------------
// M = BS = 256 (rows bs), N = NG = 256 (gate j), K = DK = 64000 (= node*32 + g)
// lstm_in generated on the fly: A[bs, n*32+g] = relu(agg[bs,n]*w_lin[g]+gb[g]).
// Grid: (4 N-strips of 64) x (50 K-chunks of 1280). Block: 512 thr = 16 waves,
// wave w owns M-rows [16w,16w+16). Partials atomicAdd'ed into gates_x.
// K loop unrolled x2 so alternate iterations use disjoint fragment registers,
// hiding the WMMA->VALU WAR hazard (4 coexec slots) behind the next loads.
#define KCHUNK 1280

__device__ inline v16h load_bfrag_f16(const float* __restrict__ row, int kb) {
  const float4* p = (const float4*)(row + kb);
  float4 q0 = p[0], q1 = p[1], q2 = p[2], q3 = p[3];
  v16h b;
  b[0]=(_Float16)q0.x; b[1]=(_Float16)q0.y; b[2]=(_Float16)q0.z; b[3]=(_Float16)q0.w;
  b[4]=(_Float16)q1.x; b[5]=(_Float16)q1.y; b[6]=(_Float16)q1.z; b[7]=(_Float16)q1.w;
  b[8]=(_Float16)q2.x; b[9]=(_Float16)q2.y; b[10]=(_Float16)q2.z; b[11]=(_Float16)q2.w;
  b[12]=(_Float16)q3.x; b[13]=(_Float16)q3.y; b[14]=(_Float16)q3.z; b[15]=(_Float16)q3.w;
  return b;
}

__global__ __launch_bounds__(512)
void k_gemm_wmma(const float* __restrict__ Wih, const float* __restrict__ wl,
                 const float* __restrict__ gb, const float* __restrict__ agg,
                 float* __restrict__ gatesx) {
  const int lane  = threadIdx.x & 31;
  const int wave  = threadIdx.x >> 5;          // 0..15
  const int mBase = wave * 16;                 // bs-row base
  const int nBase = (blockIdx.x & 3) * 64;     // gate-column base
  const int kBeg  = (blockIdx.x >> 2) * KCHUNK;
  const int kEnd  = kBeg + KCHUNK;

  // A-fragment K chunking (16-bit A 16x32): lanes<16 -> K {0..15}, lanes>=16 -> {16..31}
  const int klo   = (lane & 16) ? 16 : 0;
  const int bsRow = mBase + (lane & 15);

  // preload the 16 (w_lin, gat_bias) pairs this lane ever needs
  float wlo[8], whi[8], gbo[8], gbi[8];
  #pragma unroll
  for (int e = 0; e < 8; ++e) {
    wlo[e] = wl[klo + e];        gbo[e] = gb[klo + e];
    whi[e] = wl[8 + klo + e];   gbi[e] = gb[8 + klo + e];
  }

  // B-fragment (32x16 f16): lane holds column N=(lane&15), K = (lane>=16?16:0)+e, e=0..15 (contiguous!)
  const int kbOff = (lane & 16) ? 16 : 0;
  const float* wrow0 = Wih + (size_t)(nBase +  0 + (lane & 15)) * DK;
  const float* wrow1 = Wih + (size_t)(nBase + 16 + (lane & 15)) * DK;
  const float* wrow2 = Wih + (size_t)(nBase + 32 + (lane & 15)) * DK;
  const float* wrow3 = Wih + (size_t)(nBase + 48 + (lane & 15)) * DK;

  v8f acc0 = {}, acc1 = {}, acc2 = {}, acc3 = {};

  #pragma unroll 2
  for (int k0 = kBeg; k0 < kEnd; k0 += 32) {
    // All 32 K values of this step share node n = k0/32 -> one agg load per lane.
    const int n = k0 >> 5;
    const float a = agg[n * BS + bsRow];
    v16h af;
    #pragma unroll
    for (int e = 0; e < 8; ++e) {
      float v0 = a * wlo[e] + gbo[e];  af[e]     = (_Float16)(v0 > 0.f ? v0 : 0.f);
      float v1 = a * whi[e] + gbi[e];  af[8 + e] = (_Float16)(v1 > 0.f ? v1 : 0.f);
    }
    const int kb = k0 + kbOff;
    v16h b0 = load_bfrag_f16(wrow0, kb);
    v16h b1 = load_bfrag_f16(wrow1, kb);
    v16h b2 = load_bfrag_f16(wrow2, kb);
    v16h b3 = load_bfrag_f16(wrow3, kb);
    acc0 = __builtin_amdgcn_wmma_f32_16x16x32_f16(false, af, false, b0, (short)0, acc0, false, false);
    acc1 = __builtin_amdgcn_wmma_f32_16x16x32_f16(false, af, false, b1, (short)0, acc1, false, false);
    acc2 = __builtin_amdgcn_wmma_f32_16x16x32_f16(false, af, false, b2, (short)0, acc2, false, false);
    acc3 = __builtin_amdgcn_wmma_f32_16x16x32_f16(false, af, false, b3, (short)0, acc3, false, false);
  }

  // D layout: lane = Ncol + (M>=8 ? 16 : 0), VGPR r = M&7
  const int mrow0 = mBase + ((lane & 16) ? 8 : 0);
  const int ncol  = nBase + (lane & 15);
  #pragma unroll
  for (int r = 0; r < 8; ++r) {
    atomicAdd(&gatesx[(mrow0 + r) * NG + ncol +  0], acc0[r]);
    atomicAdd(&gatesx[(mrow0 + r) * NG + ncol + 16], acc1[r]);
    atomicAdd(&gatesx[(mrow0 + r) * NG + ncol + 32], acc2[r]);
    atomicAdd(&gatesx[(mrow0 + r) * NG + ncol + 48], acc3[r]);
  }
}

// ---------------- stage 3: sequential LSTM over S=32 (single block) ----------------
__global__ __launch_bounds__(512)
void k_lstm(const float* __restrict__ gatesx, const float* __restrict__ Whh,
            float* __restrict__ hT) {
  __shared__ float hp[BDIM * HDIM];   // 512
  __shared__ float cp[BDIM * HDIM];   // 512
  __shared__ float gt[BDIM * NG];     // 2048
  const int tid = threadIdx.x;
  hp[tid] = 0.f; cp[tid] = 0.f;
  __syncthreads();

  for (int s = 0; s < SDIM; ++s) {
    // phase 1: gates[b][j] = gates_x + hp[b] . Whh[j]   (2048 items, 4 per thread)
    #pragma unroll
    for (int p = 0; p < 4; ++p) {
      const int item = tid + p * 512;
      const int b = item >> 8, j = item & 255;
      float accv = gatesx[(b * SDIM + s) * NG + j];
      const float* wr = Whh + j * HDIM;
      const float* hb = hp + b * HDIM;
      #pragma unroll 8
      for (int k = 0; k < HDIM; ++k) accv += hb[k] * wr[k];
      gt[item] = accv;
    }
    __syncthreads();
    // phase 2: elementwise cell update, one (b,h) per thread
    {
      const int b = tid >> 6, h = tid & 63;
      const float ii = gt[b * NG + h];
      const float ff = gt[b * NG + 64 + h];
      const float gg = gt[b * NG + 128 + h];
      const float oo = gt[b * NG + 192 + h];
      float c = sigf(ff) * cp[tid] + sigf(ii) * tanhf(gg);
      cp[tid] = c;
      hp[tid] = sigf(oo) * tanhf(c);
    }
    __syncthreads();
  }
  hT[tid] = hp[tid];
}

// ---------------- stage 4: head GEMV out[b,n] = hT[b] . W_head[n] + b_head[n] ----------------
__global__ void k_head(const float* __restrict__ hT, const float* __restrict__ Whead,
                       const float* __restrict__ bhead, float* __restrict__ out) {
  int idx = blockIdx.x * 256 + threadIdx.x;
  if (idx >= BDIM * NNODE) return;
  const int b = idx / NNODE, n = idx % NNODE;
  float accv = bhead[n];
  const float* wr = Whead + n * HDIM;
  const float* hb = hT + b * HDIM;
  #pragma unroll 8
  for (int k = 0; k < HDIM; ++k) accv += hb[k] * wr[k];
  out[idx] = accv;
}

// ---------------- launch ----------------
extern "C" void kernel_launch(void* const* d_in, const int* in_sizes, int n_in,
                              void* d_out, int out_size, void* d_ws, size_t ws_size,
                              hipStream_t stream) {
  const float* x     = (const float*)d_in[0];
  const int*   ei    = (const int*)  d_in[1];
  const float* wl    = (const float*)d_in[2];
  const float* asrc  = (const float*)d_in[3];
  const float* adst  = (const float*)d_in[4];
  const float* gb    = (const float*)d_in[5];
  const float* Wih   = (const float*)d_in[6];
  const float* Whh   = (const float*)d_in[7];
  const float* bih   = (const float*)d_in[8];
  const float* bhh   = (const float*)d_in[9];
  const float* Whead = (const float*)d_in[10];
  const float* bhead = (const float*)d_in[11];
  float* out = (float*)d_out;

  // workspace layout (floats)
  float* W      = (float*)d_ws;
  float* segmax = W;                       // 512000
  float* denom  = W + 512000;              // 512000
  float* numer  = W + 1024000;             // 512000
  float* agg    = W + 1536000;             // 512000  [N][BS]
  float* gatesx = W + 2048000;             // 65536   [BS][NG]
  float* hT     = W + 2048000 + 65536;     // 512
  float* csd    = W + 2048000 + 65536 + 512; // 2

  k_scalars<<<1, 32, 0, stream>>>(wl, asrc, adst, csd);
  k_init<<<(NNODE * BS + 255) / 256, 256, 0, stream>>>(segmax, denom, numer, gatesx, bih, bhh);
  k_edge_max<<<ETOT, 256, 0, stream>>>(x, ei, csd, segmax);
  k_edge_acc<<<ETOT, 256, 0, stream>>>(x, ei, csd, segmax, denom, numer);
  k_agg<<<(NNODE * BS + 255) / 256, 256, 0, stream>>>(numer, denom, agg);
  k_gemm_wmma<<<4 * (DK / KCHUNK), 512, 0, stream>>>(Wih, wl, gb, agg, gatesx);
  k_lstm<<<1, 512, 0, stream>>>(gatesx, Whh, hT);
  k_head<<<(BDIM * NNODE + 255) / 256, 256, 0, stream>>>(hT, Whead, bhead, out);
}
